// MultiHeadSelfAttention_55336358643360
// MI455X (gfx1250) — matrix operand order repaired and checked by reading, not compiled, not run on hardware
//
#include <hip/hip_runtime.h>

#define S_LEN   4096
#define DMODEL  1024
#define NHEADS  16
#define DKH     64

typedef __bf16 bf16_t;
typedef __attribute__((ext_vector_type(16))) __bf16        v16bf;
typedef __attribute__((ext_vector_type(8)))  float         v8f;
typedef __attribute__((ext_vector_type(4)))  unsigned int  u32x4;
typedef __attribute__((ext_vector_type(8)))  int           i32x8;
typedef __attribute__((ext_vector_type(4)))  int           i32x4;

// ---- Tensor Data Mover availability (toolchain-dependent arity) -------------
#if __has_builtin(__builtin_amdgcn_tensor_load_to_lds) && \
    __has_builtin(__builtin_amdgcn_s_wait_tensorcnt)
  #define HAVE_TDM 1
  #if __has_include(<hip/amd_detail/amd_gfx1250_TDM.h>)
    #define TDM_6ARG 1          // therock headers -> clang-23 6-arg builtin
  #else
    #define TDM_6ARG 0          // ROCm 7.2 clang-22 5-arg builtin
  #endif
#else
  #define HAVE_TDM 0
#endif

__device__ __forceinline__ bf16_t f2bf(float f) { return (bf16_t)f; }  // native cvt

__device__ __forceinline__ v8f v8f_zero() {
  v8f z = {0.f, 0.f, 0.f, 0.f, 0.f, 0.f, 0.f, 0.f};
  return z;
}

__device__ __forceinline__ v8f wmma_bf16(v16bf a, v16bf b, v8f c) {
  return __builtin_amdgcn_wmma_f32_16x16x32_bf16(false, a, false, b, (short)0, c,
                                                 false, false);
}

// A fragment (16x32 bf16, MxK) from row-major [rows x ld] (contiguous k = 16*half + s).
__device__ __forceinline__ v16bf load_a_frag(const bf16_t* p, int ld, int row0,
                                             int col0, int lane) {
  const int m  = lane & 15;
  const int kb = (lane >> 4) * 16;
  const bf16_t* r = p + (size_t)(row0 + m) * ld + col0;
  v16bf a;
#pragma unroll
  for (int h = 0; h < 8; ++h) a[h] = r[kb + h];
#pragma unroll
  for (int h = 0; h < 8; ++h) a[8 + h] = r[kb + 8 + h];
  return a;
}

// B fragment (32x16 KxN) from [N][K] row-major global: elem(k,n)=p[(n0+n)*ld+k0+k]
// Per-lane contiguous (2 x b128).
__device__ __forceinline__ v16bf load_b_frag_nk(const bf16_t* p, int ld, int n0,
                                                int k0, int lane) {
  const int n  = lane & 15;
  const int kb = (lane >> 4) * 16;
  const bf16_t* r = p + (size_t)(n0 + n) * ld + k0 + kb;
  v16bf b;
#pragma unroll
  for (int h = 0; h < 16; ++h) b[h] = r[h];
  return b;
}

// B fragment from an LDS-staged [64][32] bf16 tile (row n, contiguous k)
__device__ __forceinline__ v16bf load_b_frag_lds(const bf16_t* tile, int n0,
                                                 int lane) {
  const int n  = lane & 15;
  const int kb = (lane >> 4) * 16;
  const bf16_t* r = tile + (n0 + n) * 32 + kb;
  v16bf b;
#pragma unroll
  for (int h = 0; h < 16; ++h) b[h] = r[h];
  return b;
}

#if HAVE_TDM
// Build a 2-D tensor-DMA descriptor (D#) per CDNA5 ISA 8.3/8.4 and issue
// TENSOR_LOAD_TO_LDS. Tile: tile_d0 contiguous elems x tile_d1 rows, bf16.
__device__ __forceinline__ void tdm_load_tile_2d(const bf16_t* gsrc,
                                                 unsigned lds_off,
                                                 unsigned tensor_d0,
                                                 unsigned tensor_d1,
                                                 unsigned tile_d0,
                                                 unsigned tile_d1,
                                                 unsigned stride_d0) {
  const unsigned long long ga = (unsigned long long)(size_t)gsrc;
  u32x4 g0;
  g0[0] = 1u;                                            // count=1, user D#
  g0[1] = lds_off;                                       // lds_addr (bytes)
  g0[2] = (unsigned)(ga & 0xFFFFFFFFull);                // global_addr[31:0]
  g0[3] = (unsigned)((ga >> 32) & 0x01FFFFFFull)         // global_addr[56:32]
          | (2u << 30);                                  // type=2 (image)
  i32x8 g1;
  g1[0] = (int)(1u << 16);                               // data_size=1 (2 bytes)
  g1[1] = (int)((tensor_d0 & 0xFFFFu) << 16);            // tensor_dim0[15:0]
  g1[2] = (int)((tensor_d0 >> 16) | ((tensor_d1 & 0xFFFFu) << 16));
  g1[3] = (int)((tensor_d1 >> 16) | (tile_d0 << 16));    // tile_dim0
  g1[4] = (int)(tile_d1 & 0xFFFFu);                      // tile_dim1 (dim2=0)
  g1[5] = (int)stride_d0;                                // dim0_stride[31:0]
  g1[6] = 0;                                             // stride hi / dim1_stride
  g1[7] = 0;
  i32x4 z4 = {0, 0, 0, 0};
#if TDM_6ARG
  i32x8 z8 = {0, 0, 0, 0, 0, 0, 0, 0};
  __builtin_amdgcn_tensor_load_to_lds(g0, g1, z4, z4, z8, 0);
#else
  __builtin_amdgcn_tensor_load_to_lds(g0, g1, z4, z4, 0);
#endif
}
#endif

// ---------------------------------------------------------------- conversions
__global__ void cvt_kernel(const float* __restrict__ in, bf16_t* __restrict__ out,
                           int n) {
  int i = blockIdx.x * blockDim.x + threadIdx.x;
  if (i < n) out[i] = f2bf(in[i]);
}

// ------------------------------------------------------ Q/K/V = x @ W^T (bf16)
// grid (S/256, NHEADS, 3); wave = 32x64 C tile; weight B tile (64x32) is
// DMA-staged into LDS by the Tensor Data Mover and double-buffered.
// Q/K stored per head [h][S][64]; V stored transposed [h][64][S] so that the
// attention PV B-fragments are per-lane contiguous.
__global__ __launch_bounds__(256) void qkv_gemm_kernel(
    const bf16_t* __restrict__ xb,
    const bf16_t* __restrict__ wqb, const bf16_t* __restrict__ wkb,
    const bf16_t* __restrict__ wvb,
    bf16_t* __restrict__ qh, bf16_t* __restrict__ kh, bf16_t* __restrict__ vt) {
  __shared__ bf16_t wtile[2 * 64 * 32];                  // 2 x 4KB ping-pong
  const int tid  = threadIdx.x;
  const int lane = tid & 31;
  const int wid  = tid >> 5;
  const int rw   = blockIdx.x * 256 + wid * 32;          // 32 rows per wave
  const int h    = blockIdx.y;
  const int sel  = blockIdx.z;
  const bf16_t* W   = (sel == 0) ? wqb : (sel == 1) ? wkb : wvb;
  bf16_t*       dst = (sel == 0) ? qh  : (sel == 1) ? kh  : vt;
  const float scale = (sel == 0) ? 0.125f : 1.0f;        // 1/sqrt(64) into Q
  const bf16_t* wsrc = W + (size_t)(h * 64) * DMODEL;    // 64 weight rows

  v8f acc[2][4];
#pragma unroll
  for (int mi = 0; mi < 2; ++mi)
#pragma unroll
    for (int t = 0; t < 4; ++t) acc[mi][t] = v8f_zero();

#if HAVE_TDM
  const unsigned lds0 = (unsigned)(size_t)&wtile[0];     // flat lo32 == LDS off
  const unsigned lds1 = (unsigned)(size_t)&wtile[64 * 32];
  if (wid == 0)
    tdm_load_tile_2d(wsrc, lds0, DMODEL, DMODEL, 32, 64, DMODEL);
#endif

  for (int k0 = 0; k0 < DMODEL; k0 += 32) {
    const int cur = (k0 >> 5) & 1;
    bf16_t* tile = wtile + cur * (64 * 32);
#if HAVE_TDM
    if (wid == 0) {
      if (k0 + 32 < DMODEL) {      // prefetch next tile, wait for current
        tdm_load_tile_2d(wsrc + k0 + 32, cur ? lds0 : lds1,
                         DMODEL, DMODEL, 32, 64, DMODEL);
        __builtin_amdgcn_s_wait_tensorcnt(1);
      } else {
        __builtin_amdgcn_s_wait_tensorcnt(0);
      }
    }
#else
    {                              // cooperative fallback: 256 thr x 16B = 4KB
      const int e  = tid * 8;
      const int rr = e >> 5;
      const int cc = e & 31;
      *(uint4*)(tile + e) =
          *(const uint4*)(wsrc + (size_t)rr * DMODEL + k0 + cc);
    }
#endif
    __syncthreads();

    v16bf a0 = load_a_frag(xb, DMODEL, rw,      k0, lane);
    v16bf a1 = load_a_frag(xb, DMODEL, rw + 16, k0, lane);
    if (k0 + 64 < DMODEL)
      __builtin_prefetch(xb + (size_t)(rw + (lane & 15)) * DMODEL + k0 + 64, 0, 1);
#pragma unroll
    for (int t = 0; t < 4; ++t) {
      v16bf b = load_b_frag_lds(tile, t * 16, lane);
      acc[0][t] = wmma_bf16(a0, b, acc[0][t]);
      acc[1][t] = wmma_bf16(a1, b, acc[1][t]);
    }
    __syncthreads();               // tile[cur] is rewritten next-next iter
  }

  const int cm = (lane >> 4) * 8;
  const int cn = lane & 15;
  if (sel == 2) {                  // V: transposed store vt[h][d][s]
#pragma unroll
    for (int mi = 0; mi < 2; ++mi)
#pragma unroll
      for (int t = 0; t < 4; ++t)
#pragma unroll
        for (int r = 0; r < 8; ++r) {
          const int s = rw + mi * 16 + cm + r;
          const int d = t * 16 + cn;
          dst[(size_t)h * (DKH * S_LEN) + (size_t)d * S_LEN + s] =
              f2bf(acc[mi][t][r]);
        }
  } else {                         // Q/K: per-head [h][s][d]
#pragma unroll
    for (int mi = 0; mi < 2; ++mi)
#pragma unroll
      for (int t = 0; t < 4; ++t)
#pragma unroll
        for (int r = 0; r < 8; ++r) {
          const int s = rw + mi * 16 + cm + r;
          dst[(size_t)h * (S_LEN * DKH) + (size_t)s * DKH + t * 16 + cn] =
              f2bf(acc[mi][t][r] * scale);
        }
  }
}

// ----------------------------------------------- causal flash attention (bf16)
__global__ __launch_bounds__(256) void attention_kernel(
    const bf16_t* __restrict__ qh, const bf16_t* __restrict__ kh,
    const bf16_t* __restrict__ vt, bf16_t* __restrict__ ob) {
  __shared__ bf16_t pbuf[8 * 16 * 32];                   // per-wave P staging
  const int lane = threadIdx.x & 31;
  const int wid  = threadIdx.x >> 5;
  const int qs   = blockIdx.x * 128 + wid * 16;
  const int h    = blockIdx.y;
  const bf16_t* qb  = qh + (size_t)h * S_LEN * DKH;
  const bf16_t* kbp = kh + (size_t)h * S_LEN * DKH;
  const bf16_t* vbp = vt + (size_t)h * DKH * S_LEN;      // [64][S] transposed
  bf16_t* pw = pbuf + wid * (16 * 32);
  const int cm = (lane >> 4) * 8;
  const int cn = lane & 15;

  v16bf aq0 = load_a_frag(qb, DKH, qs, 0,  lane);        // Q holds 1/sqrt(dk)
  v16bf aq1 = load_a_frag(qb, DKH, qs, 32, lane);

  const float NEG = -1.0e30f;
  float mrun[8], lsum[8];
#pragma unroll
  for (int r = 0; r < 8; ++r) { mrun[r] = NEG; lsum[r] = 0.0f; }
  v8f oacc[4] = {v8f_zero(), v8f_zero(), v8f_zero(), v8f_zero()};

  const int nb = (qs + 47) / 32;                         // causal trip count
  for (int jb = 0; jb < nb; ++jb) {
    const int j0 = jb * 32;
    if (jb + 1 < nb)
      __builtin_prefetch(kbp + (size_t)(j0 + 32 + cn) * DKH, 0, 1);
    v8f s0 = v8f_zero(), s1 = v8f_zero();
    {
      v16bf b;
      b = load_b_frag_nk(kbp, DKH, j0,      0,  lane); s0 = wmma_bf16(aq0, b, s0);
      b = load_b_frag_nk(kbp, DKH, j0,      32, lane); s0 = wmma_bf16(aq1, b, s0);
      b = load_b_frag_nk(kbp, DKH, j0 + 16, 0,  lane); s1 = wmma_bf16(aq0, b, s1);
      b = load_b_frag_nk(kbp, DKH, j0 + 16, 32, lane); s1 = wmma_bf16(aq1, b, s1);
    }
    if (j0 + 31 > qs) {                                  // diagonal block mask
#pragma unroll
      for (int r = 0; r < 8; ++r) {
        const int q = qs + cm + r;
        if (j0 + cn > q)      s0[r] = NEG;
        if (j0 + 16 + cn > q) s1[r] = NEG;
      }
    }
#pragma unroll
    for (int r = 0; r < 8; ++r) {                        // online softmax
      float v = fmaxf(s0[r], s1[r]);
      v = fmaxf(v, __shfl_xor(v, 1, 32));
      v = fmaxf(v, __shfl_xor(v, 2, 32));
      v = fmaxf(v, __shfl_xor(v, 4, 32));
      v = fmaxf(v, __shfl_xor(v, 8, 32));
      const float mnew  = fmaxf(mrun[r], v);
      const float alpha = __expf(mrun[r] - mnew);
      mrun[r] = mnew;
      const float p0 = __expf(s0[r] - mnew);
      const float p1 = __expf(s1[r] - mnew);
      pw[(cm + r) * 32 + cn]      = f2bf(p0);
      pw[(cm + r) * 32 + 16 + cn] = f2bf(p1);
      float rs = p0 + p1;
      rs += __shfl_xor(rs, 1, 32);
      rs += __shfl_xor(rs, 2, 32);
      rs += __shfl_xor(rs, 4, 32);
      rs += __shfl_xor(rs, 8, 32);
      lsum[r] = lsum[r] * alpha + rs;
#pragma unroll
      for (int t = 0; t < 4; ++t) oacc[t][r] *= alpha;
    }
    v16bf ap = load_a_frag(pw, 32, 0, 0, lane);          // P from LDS
#pragma unroll
    for (int t = 0; t < 4; ++t) {
      // B(k=key, n=d) = Vt[d][key] : [N][K] row-major, contiguous per lane
      v16bf bv = load_b_frag_nk(vbp, S_LEN, t * 16, j0, lane);
      oacc[t] = wmma_bf16(ap, bv, oacc[t]);
    }
  }

#pragma unroll
  for (int r = 0; r < 8; ++r) {
    const float inv = 1.0f / lsum[r];
    const int s = qs + cm + r;
#pragma unroll
    for (int t = 0; t < 4; ++t)
      ob[(size_t)s * DMODEL + h * DKH + t * 16 + cn] = f2bf(oacc[t][r] * inv);
  }
}

// ---------------------------------------------------- out = O @ Wo^T (f32 out)
__global__ __launch_bounds__(256) void out_proj_kernel(
    const bf16_t* __restrict__ ob, const bf16_t* __restrict__ wob,
    float* __restrict__ out) {
  __shared__ bf16_t wtile[2 * 64 * 32];
  const int tid  = threadIdx.x;
  const int lane = tid & 31;
  const int wid  = tid >> 5;
  const int rw   = blockIdx.x * 256 + wid * 32;
  const int ct   = blockIdx.y;
  const bf16_t* wsrc = wob + (size_t)(ct * 64) * DMODEL;

  v8f acc[2][4];
#pragma unroll
  for (int mi = 0; mi < 2; ++mi)
#pragma unroll
    for (int t = 0; t < 4; ++t) acc[mi][t] = v8f_zero();

#if HAVE_TDM
  const unsigned lds0 = (unsigned)(size_t)&wtile[0];
  const unsigned lds1 = (unsigned)(size_t)&wtile[64 * 32];
  if (wid == 0)
    tdm_load_tile_2d(wsrc, lds0, DMODEL, DMODEL, 32, 64, DMODEL);
#endif

  for (int k0 = 0; k0 < DMODEL; k0 += 32) {
    const int cur = (k0 >> 5) & 1;
    bf16_t* tile = wtile + cur * (64 * 32);
#if HAVE_TDM
    if (wid == 0) {
      if (k0 + 32 < DMODEL) {
        tdm_load_tile_2d(wsrc + k0 + 32, cur ? lds0 : lds1,
                         DMODEL, DMODEL, 32, 64, DMODEL);
        __builtin_amdgcn_s_wait_tensorcnt(1);
      } else {
        __builtin_amdgcn_s_wait_tensorcnt(0);
      }
    }
#else
    {
      const int e  = tid * 8;
      const int rr = e >> 5;
      const int cc = e & 31;
      *(uint4*)(tile + e) =
          *(const uint4*)(wsrc + (size_t)rr * DMODEL + k0 + cc);
    }
#endif
    __syncthreads();

    v16bf a0 = load_a_frag(ob, DMODEL, rw,      k0, lane);
    v16bf a1 = load_a_frag(ob, DMODEL, rw + 16, k0, lane);
#pragma unroll
    for (int t = 0; t < 4; ++t) {
      v16bf b = load_b_frag_lds(tile, t * 16, lane);
      acc[0][t] = wmma_bf16(a0, b, acc[0][t]);
      acc[1][t] = wmma_bf16(a1, b, acc[1][t]);
    }
    __syncthreads();
  }

  const int cm = (lane >> 4) * 8;
  const int cn = lane & 15;
#pragma unroll
  for (int mi = 0; mi < 2; ++mi)
#pragma unroll
    for (int t = 0; t < 4; ++t)
#pragma unroll
      for (int r = 0; r < 8; ++r)
        out[(size_t)(rw + mi * 16 + cm + r) * DMODEL + ct * 64 + t * 16 + cn] =
            acc[mi][t][r];
}

// ---------------------------------------------------------------------- launch
extern "C" void kernel_launch(void* const* d_in, const int* in_sizes, int n_in,
                              void* d_out, int out_size, void* d_ws, size_t ws_size,
                              hipStream_t stream) {
  (void)in_sizes; (void)n_in; (void)out_size; (void)ws_size;
  const float* x  = (const float*)d_in[0];
  const float* Wq = (const float*)d_in[1];
  const float* Wk = (const float*)d_in[2];
  const float* Wv = (const float*)d_in[3];
  const float* Wo = (const float*)d_in[4];
  float* out = (float*)d_out;

  char* ws = (char*)d_ws;
  const size_t SD = (size_t)S_LEN * DMODEL;
  const size_t DD = (size_t)DMODEL * DMODEL;
  size_t off = 0;
  bf16_t* xb  = (bf16_t*)(ws + off); off += SD * 2;
  bf16_t* wqb = (bf16_t*)(ws + off); off += DD * 2;
  bf16_t* wkb = (bf16_t*)(ws + off); off += DD * 2;
  bf16_t* wvb = (bf16_t*)(ws + off); off += DD * 2;
  bf16_t* wob = (bf16_t*)(ws + off); off += DD * 2;
  bf16_t* qh  = (bf16_t*)(ws + off); off += SD * 2;
  bf16_t* kh  = (bf16_t*)(ws + off); off += SD * 2;
  bf16_t* vt  = (bf16_t*)(ws + off); off += SD * 2;   // [H][64][S] transposed
  bf16_t* ob  = (bf16_t*)(ws + off); off += SD * 2;   // 48 MB total

  cvt_kernel<<<(int)(SD / 256), 256, 0, stream>>>(x,  xb,  (int)SD);
  cvt_kernel<<<(int)(DD / 256), 256, 0, stream>>>(Wq, wqb, (int)DD);
  cvt_kernel<<<(int)(DD / 256), 256, 0, stream>>>(Wk, wkb, (int)DD);
  cvt_kernel<<<(int)(DD / 256), 256, 0, stream>>>(Wv, wvb, (int)DD);
  cvt_kernel<<<(int)(DD / 256), 256, 0, stream>>>(Wo, wob, (int)DD);

  qkv_gemm_kernel<<<dim3(S_LEN / 256, NHEADS, 3), 256, 0, stream>>>(
      xb, wqb, wkb, wvb, qh, kh, vt);
  attention_kernel<<<dim3(S_LEN / 128, NHEADS), 256, 0, stream>>>(qh, kh, vt, ob);
  out_proj_kernel<<<dim3(S_LEN / 256, DMODEL / 64), 256, 0, stream>>>(ob, wob, out);
}
